// VectorQuantizer_74775380623864
// MI455X (gfx1250) — matrix operand order repaired and checked by reading, not compiled, not run on hardware
//
#include <hip/hip_runtime.h>

typedef unsigned short u16t;
typedef float v8f  __attribute__((ext_vector_type(8)));
typedef __bf16 v16bf __attribute__((ext_vector_type(16)));

#define B_ROWS 32768
#define K_CODES 4096
#define D_DIM 512

#define BM 128          // rows per block (8 waves x 16)
#define BN 64           // codebook entries per LDS tile
#define LDA 520         // padded LDS row stride (elements) -> conflict-free b128
#define LDB 520
#define SMEM_BYTES ((BM*LDA + BN*LDB) * 2)   // 199,680 B < 320 KB WGP LDS

// ---- ws layout (bytes) ----
#define WS_BFE   0                          // K*D bf16 codebook: 4,194,304
#define WS_ENORM 4194304                    // float[K]
#define WS_HIST  4210688                    // uint[K]
#define WS_LOSS  4227072                    // float[1] (padded)
#define WS_IDX   4227088                    // uint[B]

__device__ __forceinline__ u16t f2bf(float f) {
    unsigned u = __float_as_uint(f);
    unsigned r = u + 0x7FFFu + ((u >> 16) & 1u);   // round-to-nearest-even
    return (u16t)(r >> 16);
}

union FragU { v16bf v; uint4 q[2]; };

// Kernel 0: bf16 codebook + ||e_k||^2 + zero hist/lossSum. grid=K, block=128.
__global__ void vq_prep(const float* __restrict__ E, u16t* __restrict__ bfE,
                        float* __restrict__ eNorm, unsigned* __restrict__ hist,
                        float* __restrict__ lossSum) {
    const int k = blockIdx.x, t = threadIdx.x;
    float4 v = ((const float4*)(E + (size_t)k * D_DIM))[t];
    uint2 p;
    p.x = (unsigned)f2bf(v.x) | ((unsigned)f2bf(v.y) << 16);
    p.y = (unsigned)f2bf(v.z) | ((unsigned)f2bf(v.w) << 16);
    ((uint2*)(bfE + (size_t)k * D_DIM))[t] = p;
    float s = v.x*v.x + v.y*v.y + v.z*v.z + v.w*v.w;
#pragma unroll
    for (int m = 16; m >= 1; m >>= 1) s += __shfl_xor(s, m, 32);
    __shared__ float wsum[4];
    if ((t & 31) == 0) wsum[t >> 5] = s;
    __syncthreads();
    if (t == 0) {
        eNorm[k] = wsum[0] + wsum[1] + wsum[2] + wsum[3];
        hist[k] = 0u;
        if (k == 0) lossSum[0] = 0.0f;
    }
}

// Kernel 1: zero the one-hot encodings output (B*K floats), float2 stores.
__global__ void vq_zero_enc(float2* __restrict__ enc2, long long n2) {
    long long i = (long long)blockIdx.x * blockDim.x + threadIdx.x;
    long long stride = (long long)gridDim.x * blockDim.x;
    for (; i < n2; i += stride) enc2[i] = make_float2(0.0f, 0.0f);
}

// Kernel 2: fused bf16 WMMA GEMM (X * E^T) + row argmin of ||e||^2 - 2*dot.
__global__ __launch_bounds__(256)
void vq_argmin_wmma(const float* __restrict__ X, const u16t* __restrict__ bfE,
                    const float* __restrict__ eNorm,
                    unsigned* __restrict__ idxBuf, float* __restrict__ outIdx) {
    extern __shared__ char smem[];
    u16t* sA = (u16t*)smem;                    // [BM][LDA] bf16
    u16t* sB = (u16t*)(smem + BM * LDA * 2);   // [BN][LDB] bf16

    const int tid  = threadIdx.x;
    const int w    = tid >> 5;
    const int lane = tid & 31;
    const int m    = lane & 15;     // M (A row) or N (B/C col) sub-index
    const int hi   = lane >> 4;     // half-wave select
    const int rowBase = blockIdx.x * BM;

    // Stage A tile: 128x512 f32 -> bf16 in LDS (one-time per block).
    for (int i = tid; i < BM * (D_DIM / 4); i += 256) {
        int r = i >> 7, c4 = i & 127;
        float4 v = ((const float4*)(X + (size_t)(rowBase + r) * D_DIM))[c4];
        uint2 p;
        p.x = (unsigned)f2bf(v.x) | ((unsigned)f2bf(v.y) << 16);
        p.y = (unsigned)f2bf(v.z) | ((unsigned)f2bf(v.w) << 16);
        *(uint2*)(sA + r * LDA + c4 * 4) = p;
    }
    __syncthreads();

    float    best[8];
    unsigned bidx[8];
#pragma unroll
    for (int r = 0; r < 8; ++r) { best[r] = 3.4e38f; bidx[r] = 0u; }

    for (int kt = 0; kt < K_CODES / BN; ++kt) {
        const int nBase = kt * BN;
        // Stage B tile: 64x512 bf16 straight copy from L2-resident bf16 codebook.
        for (int i = tid; i < BN * (D_DIM / 8); i += 256) {
            int r = i >> 6, c8 = i & 63;
            uint4 v = ((const uint4*)(bfE + (size_t)(nBase + r) * D_DIM))[c8];
            *(uint4*)(sB + r * LDB + c8 * 8) = v;
        }
        __syncthreads();

        const u16t* aRow = sA + (16 * w + m) * LDA;
#pragma unroll
        for (int nSub = 0; nSub < 4; ++nSub) {
            v8f acc = {0.f, 0.f, 0.f, 0.f, 0.f, 0.f, 0.f, 0.f};
            const u16t* bRow = sB + (nSub * 16 + m) * LDB;
#pragma unroll
            for (int dstep = 0; dstep < D_DIM / 32; ++dstep) {
                const int kb = dstep * 32;
                FragU a, b;
                // A 16x32 bf16 layout: lanes0-15 K=0..7,16..23 ; lanes16-31 K=8..15,24..31
                a.q[0] = *(const uint4*)(aRow + kb + 8 * hi);
                a.q[1] = *(const uint4*)(aRow + kb + 8 * hi + 16);
                // B 32x16 bf16 layout: lanes0-15 K=0..7,16..23 ; lanes16-31 K=8..15,24..31
                b.q[0] = *(const uint4*)(bRow + kb + 8 * hi);
                b.q[1] = *(const uint4*)(bRow + kb + 8 * hi + 16);
                acc = __builtin_amdgcn_wmma_f32_16x16x32_bf16(
                        false, a.v, false, b.v, (short)0, acc, false, false);
            }
            // Epilogue: C layout lane=N, VGPR r -> M = r + 8*hi. argmin update.
            const int nCol = nBase + nSub * 16 + m;
            const float en = eNorm[nCol];
#pragma unroll
            for (int r = 0; r < 8; ++r) {
                float dist = en - 2.0f * acc[r];
                if (dist < best[r]) { best[r] = dist; bidx[r] = (unsigned)nCol; }
            }
        }
        __syncthreads();
    }

    // Reduce (val,idx) across the 16 lanes sharing each output row; prefer low idx.
#pragma unroll
    for (int r = 0; r < 8; ++r) {
        float v = best[r]; unsigned i = bidx[r];
#pragma unroll
        for (int mk = 8; mk >= 1; mk >>= 1) {
            float    ov = __shfl_xor(v, mk, 32);
            unsigned oi = (unsigned)__shfl_xor((int)i, mk, 32);
            if (ov < v || (ov == v && oi < i)) { v = ov; i = oi; }
        }
        best[r] = v; bidx[r] = i;
    }
    if (m == 0) {
#pragma unroll
        for (int r = 0; r < 8; ++r) {
            int gRow = rowBase + 16 * w + 8 * hi + r;
            idxBuf[gRow] = bidx[r];
            outIdx[gRow] = (float)bidx[r];
        }
    }
}

// Kernel 3: gather quantized rows (exact f32 codebook), loss partials,
// histogram, one-hot scatter. grid=B, block=128.
__global__ void vq_gather_loss(const float* __restrict__ X, const float* __restrict__ E,
                               const unsigned* __restrict__ idxBuf,
                               float* __restrict__ outQ, float* __restrict__ outEnc,
                               unsigned* __restrict__ hist, float* __restrict__ lossSum) {
    const int row = blockIdx.x, t = threadIdx.x;
    const unsigned idx = idxBuf[row];
    float4 q4 = ((const float4*)(E + (size_t)idx * D_DIM))[t];
    float4 x4 = ((const float4*)(X + (size_t)row * D_DIM))[t];
    ((float4*)(outQ + (size_t)row * D_DIM))[t] = q4;   // quantized_st == quantized
    float dx = q4.x - x4.x, dy = q4.y - x4.y, dz = q4.z - x4.z, dw = q4.w - x4.w;
    float s = dx*dx + dy*dy + dz*dz + dw*dw;
#pragma unroll
    for (int m = 16; m >= 1; m >>= 1) s += __shfl_xor(s, m, 32);
    if ((t & 31) == 0) atomicAdd(lossSum, s);
    if (t == 0) {
        atomicAdd(hist + idx, 1u);
        outEnc[(size_t)row * K_CODES + idx] = 1.0f;
    }
}

// Kernel 4: loss scale + perplexity. 1 block.
__global__ void vq_finalize(const unsigned* __restrict__ hist,
                            const float* __restrict__ lossSum,
                            float* __restrict__ outLoss, float* __restrict__ outPerp) {
    __shared__ float red[256];
    const int t = threadIdx.x;
    float s = 0.0f;
    for (int k = t; k < K_CODES; k += 256) {
        float p = (float)hist[k] * (1.0f / (float)B_ROWS);
        s += p * __logf(p + 1e-10f);
    }
    red[t] = s;
    __syncthreads();
    for (int off = 128; off > 0; off >>= 1) {
        if (t < off) red[t] += red[t + off];
        __syncthreads();
    }
    if (t == 0) {
        *outPerp = __expf(-red[0]);
        *outLoss = 1.25f * lossSum[0] / (float)((size_t)B_ROWS * D_DIM);
    }
}

extern "C" void kernel_launch(void* const* d_in, const int* in_sizes, int n_in,
                              void* d_out, int out_size, void* d_ws, size_t ws_size,
                              hipStream_t stream) {
    const float* X = (const float*)d_in[0];   // [B, D] f32
    const float* E = (const float*)d_in[1];   // [K, D] f32
    float* out = (float*)d_out;

    // Output layout (tuple concatenated, all f32):
    const size_t OFF_IDX  = (size_t)B_ROWS * D_DIM;   // 16,777,216
    const size_t OFF_LOSS = OFF_IDX + B_ROWS;
    const size_t OFF_PERP = OFF_LOSS + 1;
    const size_t OFF_ENC  = OFF_PERP + 1;

    char* ws = (char*)d_ws;
    u16t*     bfE     = (u16t*)(ws + WS_BFE);
    float*    eNorm   = (float*)(ws + WS_ENORM);
    unsigned* hist    = (unsigned*)(ws + WS_HIST);
    float*    lossSum = (float*)(ws + WS_LOSS);
    unsigned* idxBuf  = (unsigned*)(ws + WS_IDX);

    vq_prep<<<K_CODES, 128, 0, stream>>>(E, bfE, eNorm, hist, lossSum);

    vq_zero_enc<<<4096, 256, 0, stream>>>((float2*)(out + OFF_ENC),
                                          (long long)B_ROWS * K_CODES / 2);

    (void)hipFuncSetAttribute((const void*)vq_argmin_wmma,
                              hipFuncAttributeMaxDynamicSharedMemorySize, SMEM_BYTES);
    vq_argmin_wmma<<<B_ROWS / BM, 256, SMEM_BYTES, stream>>>(X, bfE, eNorm,
                                                             idxBuf, out + OFF_IDX);

    vq_gather_loss<<<B_ROWS, 128, 0, stream>>>(X, E, idxBuf,
                                               out, out + OFF_ENC, hist, lossSum);

    vq_finalize<<<1, 256, 0, stream>>>(hist, lossSum, out + OFF_LOSS, out + OFF_PERP);
}
